// SearchTransformer_16234976379214
// MI455X (gfx1250) — matrix operand order repaired and checked by reading, not compiled, not run on hardware
//
#include <hip/hip_runtime.h>

// ---------------------------------------------------------------------------
// SearchTransformer for MI455X (gfx1250, wave32, WMMA + Tensor Data Mover)
//
//  (0) memset packed (max,argmax) accumulator
//  (1) norm_pack : build L2-normalized f16 patch matrices Q,K  [pos][2304]
//  (2) gemm_argmax : K^T*Q via v_wmma_f32_16x16x32_f16; key panels staged
//      into LDS by TDM (tensor_load_to_lds), double buffered, TENSORcnt sync
//  (3) finalize   : decode packed -> soft_attention + max_index
//  (4) tex_kernel x3 : fused unfold->gather->fold overlap-add per level
// ---------------------------------------------------------------------------

typedef __attribute__((ext_vector_type(16))) _Float16 v16h;
typedef __attribute__((ext_vector_type(8)))  _Float16 v8h;
typedef __attribute__((ext_vector_type(8)))  float    v8f;
typedef __attribute__((ext_vector_type(4)))  unsigned int v4u;
typedef __attribute__((ext_vector_type(8)))  int      v8i;
typedef __attribute__((ext_vector_type(4)))  int      v4i;

#define KDIM   2304      // C*9 reduction dimension
#define DCHUNK 128       // K-dim staging chunk (64 DWORD rows -> TDM pad ok)
#define LDA    136       // padded LDS row stride in halfs (128 + 8)
#define NPOS   4096      // 64*64 spatial positions
#define NBATCH 2
#define NCHUNK (KDIM / DCHUNK)          // 18 chunks per key group
#define NKG    16                       // 16 key groups of 64 per slice

#if __has_builtin(__builtin_amdgcn_tensor_load_to_lds)
#define HAVE_TDM 1
#else
#define HAVE_TDM 0
#endif

// ---- order-preserving float <-> u32 encoding (for packed u64 atomicMax) ----
__device__ __forceinline__ unsigned enc_f(float f) {
  unsigned b = __float_as_uint(f);
  return (b & 0x80000000u) ? ~b : (b | 0x80000000u);
}

__device__ __forceinline__ v16h cat8(v8h lo, v8h hi) {
  return __builtin_shufflevector(lo, hi, 0,1,2,3,4,5,6,7,8,9,10,11,12,13,14,15);
}

// reduce one 16x16 f32 C-tile into running packed (val,key) max per lane.
// C layout: lanes 0-15 hold N=0-15 / M=r ; lanes 16-31 hold N=0-15 / M=r+8.
__device__ __forceinline__ void tile_reduce(const v8f& c, int keybase, int lane,
                                            unsigned long long& best) {
  int moff = (lane & 16) ? 8 : 0;
#pragma unroll
  for (int r = 0; r < 8; ++r) {
    unsigned long long p = ((unsigned long long)enc_f(c[r]) << 32)
                         | (unsigned)(keybase + moff + r);
    best = (p > best) ? p : best;
  }
}

#if HAVE_TDM
// Issue a TDM load of a 64-row x 128-half f16 tile (global row stride KDIM)
// into LDS at byte offset lds_off, with LDS row padding 64 DWORDs data +
// 4 DWORDs pad (row stride = 136 halfs = LDA). D# per CDNA5 ISA ch.8.
// This toolchain's builtin takes 6 args: D# groups 0-3 (2-D tile: groups
// 2/3 zero), an extra int32x8 descriptor word block (zero), and cpol.
__device__ __forceinline__ void tdm_load_tile(unsigned lds_off,
                                              const _Float16* gptr) {
  unsigned long long ga = (unsigned long long)(size_t)gptr;
  v4u g0;
  g0[0] = 1u;                                   // count=1, user descriptor
  g0[1] = lds_off;                              // lds_addr (bytes)
  g0[2] = (unsigned)ga;                         // global_addr[31:0]
  g0[3] = (unsigned)((ga >> 32) & 0x01FFFFFFu)  // global_addr[56:32]
        | (2u << 30);                           // type = 2 ("image")
  v8i g1;
  g1[0] = (int)((1u << 16)                      // data_size = 2 bytes
              | (1u << 20)                      // pad_enable
              | (5u << 22)                      // pad_interval: 64 DWORDs
              | (3u << 25));                    // pad_amount : 4 DWORDs
  g1[1] = (int)((unsigned)DCHUNK << 16);        // tensor_dim0 lo16 @ [63:48]
  g1[2] = (int)((unsigned)64 << 16);            // dim0 hi16=0 | tensor_dim1=64
  g1[3] = (int)((unsigned)DCHUNK << 16);        // dim1 hi16=0 | tile_dim0=128
  g1[4] = (int)64u;                             // tile_dim1=64, tile_dim2=0
  g1[5] = (int)(unsigned)KDIM;                  // tensor_dim0_stride lo32
  g1[6] = 0;                                    // stride hi16 | dim1_stride lo
  g1[7] = 0;
  v4i z4 = {};
  v8i z8 = {};
  __builtin_amdgcn_tensor_load_to_lds(g0, g1, z4, z4, z8, 0);
}
#endif

// ---------------------------------------------------------------------------
// (1) Build normalized f16 patch rows. One block per (image, batch, position).
// ---------------------------------------------------------------------------
__global__ __launch_bounds__(256) void norm_pack(
    const float* __restrict__ lq, const float* __restrict__ rdu,
    _Float16* __restrict__ Qn, _Float16* __restrict__ Kn) {
  int b     = blockIdx.x;
  int pos   = b & (NPOS - 1);
  int n     = (b >> 12) & 1;
  int isKey = b >> 13;
  const float* src = isKey ? rdu : lq;
  _Float16*    dst = isKey ? Kn  : Qn;

  int y = pos >> 6, x = pos & 63;
  int t = threadIdx.x;

  float vals[9];
  float ss = 0.f;
#pragma unroll
  for (int i = 0; i < 9; ++i) {
    int d  = t + i * 256;           // 0..2303
    int c  = d / 9;
    int j  = d - c * 9;
    int yy = y + j / 3 - 1;
    int xx = x + (j % 3) - 1;
    float v = 0.f;
    if ((unsigned)yy < 64u && (unsigned)xx < 64u)
      v = src[(((size_t)n * 256 + c) << 12) + (yy << 6) + xx];
    vals[i] = v;
    ss += v * v;
  }

  __shared__ float red[256];
  red[t] = ss;
  __syncthreads();
  for (int off = 128; off > 0; off >>= 1) {
    if (t < off) red[t] += red[t + off];
    __syncthreads();
  }
  float inv = 1.0f / fmaxf(sqrtf(red[0]), 1e-12f);

  _Float16* drow = dst + ((size_t)n * NPOS + pos) * KDIM;
#pragma unroll
  for (int i = 0; i < 9; ++i)
    drow[t + i * 256] = (_Float16)(vals[i] * inv);
}

// ---------------------------------------------------------------------------
// (2) rel = K^T * Q with streaming max/argmax.
// Grid: 2(batch) x 32(q-blocks of 128) x 4(key slices of 1024) = 256 blocks.
// Block: 256 threads = 8 waves; wave w owns queries [qb*128+w*16, +16).
// Wave computes 64(key) x 16(query) per pass: 4 WMMA accum tiles.
// Key panels DMA'd into double-buffered LDS by the Tensor Data Mover,
// overlapped with WMMA compute; one barrier per chunk (wave0 waits
// TENSORcnt before arriving, so barrier release implies data ready).
// ---------------------------------------------------------------------------
__global__ __launch_bounds__(256, 1) void gemm_argmax(
    const _Float16* __restrict__ Kn, const _Float16* __restrict__ Qn,
    unsigned long long* __restrict__ packed) {
  __shared__ _Float16 ldsA[2][64 * LDA];   // 2 x 17,408 B

  int wg   = blockIdx.x;
  int ks   = wg & 3;            // key slice (1024 keys)
  int qb   = (wg >> 2) & 31;    // query block (128 queries)
  int n    = wg >> 7;           // batch
  int lane = threadIdx.x & 31;
  int wave = threadIdx.x >> 5;

  int qbase = qb * 128 + wave * 16;
  // B fragment: lane<16 -> query qbase+lane, dims {d..d+7, d+16..d+23}; lane>=16 -> +8
  const _Float16* Qrow =
      Qn + ((size_t)n * NPOS + qbase + (lane & 15)) * KDIM + ((lane >> 4) << 3);
  const _Float16* Kbase = Kn + ((size_t)n * NPOS + ks * 1024) * KDIM;

  int aoff = (lane & 16) ? 8 : 0;   // A frag: dims {d..d+7, d+16..d+23} (+8 hi)

#if HAVE_TDM
  unsigned ldsoff[2] = { (unsigned)(size_t)&ldsA[0][0],
                         (unsigned)(size_t)&ldsA[1][0] };
  bool issuer = (threadIdx.x < 32);
  if (issuer) tdm_load_tile(ldsoff[0], Kbase);   // chunk 0 -> buf 0
#endif

  unsigned long long best = 0;  // encodes -inf
  v8f c0 = {}, c1 = {}, c2 = {}, c3 = {};
  int kg = 0, dsub = 0;

  for (int t = 0; t < NKG * NCHUNK; ++t) {
#if HAVE_TDM
    if (issuer) __builtin_amdgcn_s_wait_tensorcnt(0);
    __syncthreads();
    {  // kick DMA for chunk t+1 into the other buffer (freed by this barrier)
      int ndsub = dsub + 1, nkg = kg;
      if (ndsub == NCHUNK) { ndsub = 0; nkg = kg + 1; }
      if (issuer && nkg < NKG)
        tdm_load_tile(ldsoff[(t + 1) & 1],
                      Kbase + (size_t)nkg * 64 * KDIM + ndsub * DCHUNK);
    }
    const _Float16* arow = &ldsA[t & 1][(lane & 15) * LDA];
#else
    __syncthreads();
    {  // manual staging fallback: 64 x 128 halfs, 4 x 16B per thread
      const _Float16* gsrc = Kbase + (size_t)kg * 64 * KDIM + dsub * DCHUNK;
#pragma unroll
      for (int j = 0; j < 4; ++j) {
        int chunk = threadIdx.x + j * 256;
        int row   = chunk >> 4;
        int col   = (chunk & 15) << 3;
        *(v8h*)(&ldsA[0][row * LDA + col]) =
            *(const v8h*)(gsrc + (size_t)row * KDIM + col);
      }
    }
    __syncthreads();
    const _Float16* arow = &ldsA[0][(lane & 15) * LDA];
#endif

    int dcg = dsub * DCHUNK;
    if (dsub + 1 < NCHUNK)
      __builtin_prefetch(Qrow + dcg + DCHUNK, 0, 0);  // next Q panel (L2)

#pragma unroll
    for (int d0 = 0; d0 < DCHUNK; d0 += 32) {
      v16h bf = cat8(*(const v8h*)(Qrow + dcg + d0),
                     *(const v8h*)(Qrow + dcg + d0 + 16));
      int dlo = d0 + aoff;
      v16h a;
      a  = cat8(*(const v8h*)(arow + 0 * 16 * LDA + dlo),
                *(const v8h*)(arow + 0 * 16 * LDA + dlo + 16));
      c0 = __builtin_amdgcn_wmma_f32_16x16x32_f16(false, a, false, bf,
                                                  (short)0, c0, false, false);
      a  = cat8(*(const v8h*)(arow + 1 * 16 * LDA + dlo),
                *(const v8h*)(arow + 1 * 16 * LDA + dlo + 16));
      c1 = __builtin_amdgcn_wmma_f32_16x16x32_f16(false, a, false, bf,
                                                  (short)0, c1, false, false);
      a  = cat8(*(const v8h*)(arow + 2 * 16 * LDA + dlo),
                *(const v8h*)(arow + 2 * 16 * LDA + dlo + 16));
      c2 = __builtin_amdgcn_wmma_f32_16x16x32_f16(false, a, false, bf,
                                                  (short)0, c2, false, false);
      a  = cat8(*(const v8h*)(arow + 3 * 16 * LDA + dlo),
                *(const v8h*)(arow + 3 * 16 * LDA + dlo + 16));
      c3 = __builtin_amdgcn_wmma_f32_16x16x32_f16(false, a, false, bf,
                                                  (short)0, c3, false, false);
    }

    if (dsub == NCHUNK - 1) {   // finished full K-dim for this key group
      int kgbase = ks * 1024 + kg * 64;
      tile_reduce(c0, kgbase +  0, lane, best);
      tile_reduce(c1, kgbase + 16, lane, best);
      tile_reduce(c2, kgbase + 32, lane, best);
      tile_reduce(c3, kgbase + 48, lane, best);
      c0 = (v8f){}; c1 = (v8f){}; c2 = (v8f){}; c3 = (v8f){};
      kg += 1; dsub = 0;
    } else {
      dsub += 1;
    }
  }

  // combine lane pairs (L, L+16) which hold the same query column
  unsigned lo = (unsigned)best, hi = (unsigned)(best >> 32);
  lo = __shfl_xor(lo, 16, 32);
  hi = __shfl_xor(hi, 16, 32);
  unsigned long long other = ((unsigned long long)hi << 32) | lo;
  if (other > best) best = other;

  if (lane < 16)
    atomicMax(packed + (size_t)n * NPOS + qbase + lane, best);
}

// ---------------------------------------------------------------------------
// (3) decode packed (max,argmax) -> soft_attention + index array
// ---------------------------------------------------------------------------
__global__ void finalize_k(const unsigned long long* __restrict__ packed,
                           float* __restrict__ soft, int* __restrict__ midx) {
  int i = blockIdx.x * blockDim.x + threadIdx.x;
  if (i < NBATCH * NPOS) {
    unsigned long long p = packed[i];
    unsigned e = (unsigned)(p >> 32);
    unsigned b = (e & 0x80000000u) ? (e & 0x7FFFFFFFu) : ~e;
    soft[i] = __uint_as_float(b);
    midx[i] = (int)(p & 0xFFFFFFFFu);
  }
}

// ---------------------------------------------------------------------------
// (4) texture = fold(gather(unfold(ref)))/9, fused overlap-add gather.
// ---------------------------------------------------------------------------
__global__ void tex_kernel(const float* __restrict__ ref,
                           const int* __restrict__ midx,
                           float* __restrict__ out,
                           int C, int H, int W, int s) {
  long long total  = (long long)NBATCH * C * H * W;
  long long stride = (long long)gridDim.x * blockDim.x;
  for (long long i = blockIdx.x * (long long)blockDim.x + threadIdx.x;
       i < total; i += stride) {
    int x = (int)(i % W);
    long long r = i / W;
    int y = (int)(r % H); r /= H;
    int c = (int)(r % C);
    int n = (int)(r / C);
    int fy = y / s, fx = x / s;
    float acc = 0.f;
    for (int ty = fy - 2; ty <= fy; ++ty) {
      int qy = ty + 1;
      if ((unsigned)qy > 63u) continue;
      int dy = y - s * ty;                    // in-window row offset [0,3s)
      for (int tx = fx - 2; tx <= fx; ++tx) {
        int qx = tx + 1;
        if ((unsigned)qx > 63u) continue;
        int dx = x - s * tx;
        int m  = midx[(size_t)n * NPOS + qy * 64 + qx];
        int ry = s * ((m >> 6) - 1) + dy;     // matched window start + offset
        int rx = s * ((m & 63) - 1) + dx;
        if ((unsigned)ry < (unsigned)H && (unsigned)rx < (unsigned)W)
          acc += ref[(((size_t)n * C + c) * H + ry) * W + rx];
      }
    }
    out[i] = acc * (1.0f / 9.0f);
  }
}

// ---------------------------------------------------------------------------
extern "C" void kernel_launch(void* const* d_in, const int* in_sizes, int n_in,
                              void* d_out, int out_size, void* d_ws, size_t ws_size,
                              hipStream_t stream) {
  (void)in_sizes; (void)n_in; (void)out_size; (void)ws_size;
  const float* lq   = (const float*)d_in[0];   // (2,256,64,64)
  const float* rdu  = (const float*)d_in[1];   // (2,256,64,64)
  const float* ref0 = (const float*)d_in[2];   // (2,256,64,64)
  const float* ref1 = (const float*)d_in[3];   // (2,128,128,128)
  const float* ref2 = (const float*)d_in[4];   // (2,64,256,256)

  char* ws = (char*)d_ws;
  const size_t PACKSZ = (size_t)NBATCH * NPOS * KDIM * sizeof(_Float16); // 37.75 MB
  _Float16* Kn = (_Float16*)ws;
  _Float16* Qn = (_Float16*)(ws + PACKSZ);
  unsigned long long* packed =
      (unsigned long long*)(ws + 2 * PACKSZ);
  int* midx = (int*)(ws + 2 * PACKSZ + (size_t)NBATCH * NPOS * sizeof(unsigned long long));

  float* out  = (float*)d_out;
  float* soft = out;                                   // (2,1,64,64)
  float* t0   = soft + (size_t)NBATCH * NPOS;          // (2,256,64,64)
  float* t1   = t0 + (size_t)NBATCH * 256 * 64 * 64;   // (2,128,128,128)
  float* t2   = t1 + (size_t)NBATCH * 128 * 128 * 128; // (2,64,256,256)

  (void)hipMemsetAsync(packed, 0,
                       (size_t)NBATCH * NPOS * sizeof(unsigned long long),
                       stream);

  norm_pack<<<2 * NBATCH * NPOS, 256, 0, stream>>>(lq, rdu, Qn, Kn);
  gemm_argmax<<<256, 256, 0, stream>>>(Kn, Qn, packed);
  finalize_k<<<(NBATCH * NPOS + 255) / 256, 256, 0, stream>>>(packed, soft, midx);

  long long n0 = (long long)NBATCH * 256 * 64 * 64;
  long long n1 = (long long)NBATCH * 128 * 128 * 128;
  long long n2 = (long long)NBATCH * 64 * 256 * 256;
  tex_kernel<<<(unsigned)((n0 + 255) / 256), 256, 0, stream>>>(ref0, midx, t0, 256, 64, 64, 1);
  tex_kernel<<<(unsigned)((n1 + 255) / 256), 256, 0, stream>>>(ref1, midx, t1, 128, 128, 128, 2);
  tex_kernel<<<(unsigned)((n2 + 255) / 256), 256, 0, stream>>>(ref2, midx, t2, 64, 256, 256, 4);
}
